// PositionAwareARCEncoder_16466904613270
// MI455X (gfx1250) — compile-verified, matrix operand order checked by reading, not run on hardware
//
#include <hip/hip_runtime.h>
#include <hip/hip_bf16.h>

// ---------------------------------------------------------------------------
// MI455X (gfx1250) fused ARC encoder.
// Roofline: ~98 GFLOP total vs ~26MB HBM traffic -> compute bound.
// Strategy: one workgroup per 16-batch tile; every GEMM runs on
// v_wmma_f32_16x16x32_bf16 (f32 accumulate); all intermediates in LDS;
// weights pre-packed into ISA B-fragment lane layout so main-kernel weight
// loads are two global_load_b128 per fragment; flash-style online softmax
// in registers (threads 0..63 own one (batch,head) pair).
// ---------------------------------------------------------------------------

typedef __bf16 bf16_t;
typedef __attribute__((ext_vector_type(16))) __bf16 v16bf;
typedef __attribute__((ext_vector_type(8)))  __bf16 v8bf;
typedef __attribute__((ext_vector_type(8)))  float  v8f;

#define DEVINL static __device__ __forceinline__

constexpr int TB = 16;   // batch tile per workgroup
constexpr int P  = 25;   // positions (5x5)

// workspace offsets in bf16 elements (all multiples of 8 -> 16B aligned)
constexpr size_t OFF_CE   = 0;       // color_embed     4*64
constexpr size_t OFF_POS  = 256;     // pos table       25*64
constexpr size_t OFF_IPW1 = 1856;    // ip_w1 frags     8 ntiles * 4 kchunks * 512
constexpr size_t OFF_IPW2 = 18240;   // ip_w2           4*4*512
constexpr size_t OFF_OPW1 = 26432;   // op_w1           8*4*512
constexpr size_t OFF_OPW2 = 42816;   // op_w2           4*4*512
constexpr size_t OFF_QW   = 51008;   // q_w             4*2*512
constexpr size_t OFF_KW   = 55104;   // k_w             4*2*512
constexpr size_t OFF_VW   = 59200;   // v_w             4*2*512
constexpr size_t OFF_OW   = 63296;   // o_w             4*2*512
constexpr size_t OFF_CBW1 = 67392;   // cb_w1           8*6*512
constexpr size_t OFF_CBW2 = 91968;   // cb_w2           8*4*512   (end 108352 elems ~212KB)

union VU { v16bf v; v8bf h[2]; };

DEVINL v8f wmma_bf16(v16bf a, v16bf b, v8f c) {
  // 8 args: (neg_a, A, neg_b, B, c_mod, C, reuse_a, reuse_b)
  return __builtin_amdgcn_wmma_f32_16x16x32_bf16(false, a, false, b, (short)0, c,
                                                 false, false);
}

// B-matrix fragment: pre-packed per-lane, 16 contiguous bf16 per lane.
DEVINL v16bf bfrag(const bf16_t* pk, int frag, int lane) {
  const bf16_t* p = pk + (((size_t)frag << 5) + lane) * 16;
  VU u;
  u.h[0] = *(const v8bf*)(p);
  u.h[1] = *(const v8bf*)(p + 8);
  return u.v;
}

// A-matrix fragment from row-major bf16 LDS tile (stride: multiple of 8).
// ISA 16-bit A 16x32 layout: lane row = lane&15; lanes>=16 take K+8 offset;
// elements 0..7 = K k0..k0+7, elements 8..15 = K k0+16..k0+23.
DEVINL v16bf afrag_lds(const bf16_t* buf, int stride, int kbase, int lane) {
  int row = lane & 15, hf = lane >> 4;
  const bf16_t* p = buf + row * stride + kbase + hf * 8;
  VU u;
  u.h[0] = *(const v8bf*)(p);
  u.h[1] = *(const v8bf*)(p + 16);
  return u.v;
}

// Same, from an f32 LDS tile (converted on the fly).
DEVINL v16bf afrag_f32(const float* buf, int stride, int kbase, int lane) {
  int row = lane & 15, hf = lane >> 4;
  const float* p = buf + row * stride + kbase + hf * 8;
  v16bf a;
#pragma unroll
  for (int i = 0; i < 8; i++) a[i] = (bf16_t)p[i];
#pragma unroll
  for (int i = 0; i < 8; i++) a[8 + i] = (bf16_t)p[16 + i];
  return a;
}

// 8 contiguous features of in_feat/out_feat: [color_embed[g] (64) | pos[p] (64)]
DEVINL v8bf feat8(const bf16_t* ce_g, const bf16_t* pos_p, int idx) {
  const bf16_t* src = (idx < 64) ? (ce_g + idx) : (pos_p + (idx - 64));
  return *(const v8bf*)src;
}

DEVINL v16bf afrag_feat(const bf16_t* ce_g, const bf16_t* pos_p, int kbase, int hf) {
  int i0 = kbase + hf * 8;
  VU u;
  u.h[0] = feat8(ce_g, pos_p, i0);
  u.h[1] = feat8(ce_g, pos_p, i0 + 16);
  return u.v;
}

// ---------------------------------------------------------------------------
// Packing kernels
// ---------------------------------------------------------------------------

// Pack weight W[K][N] (row-major, fan_in x fan_out) into WMMA B-fragments.
// grid.x = ntiles*kchunks, block = 512 (32 lanes x 16 elements).
__global__ void pack_wmma_b(const float* __restrict__ w, bf16_t* __restrict__ dst,
                            int N, int kchunks) {
  int frag = blockIdx.x;
  int nt = frag / kchunks, kc = frag % kchunks;
  int lane = threadIdx.x & 31, e = threadIdx.x >> 5;
  int hf = lane >> 4;
  int n = nt * 16 + (lane & 15);
  int k = kc * 32 + ((e < 8) ? 0 : 16) + hf * 8 + (e & 7);
  dst[((size_t)frag * 32 + lane) * 16 + e] = (bf16_t)w[(size_t)k * N + n];
}

// bf16 copies of color_embed and the combined positional table.
__global__ void pack_embeds(const float* __restrict__ ce, const float* __restrict__ re,
                            const float* __restrict__ co, bf16_t* __restrict__ ws) {
  int t = blockIdx.x * blockDim.x + threadIdx.x;
  if (t < 4 * 64) ws[OFF_CE + t] = (bf16_t)ce[t];
  if (t < 25 * 64) {
    int p = t >> 6, j = t & 63;
    int r = p / 5, c = p % 5;
    float v = (j < 32) ? re[r * 32 + j] : co[c * 32 + (j - 32)];
    ws[OFF_POS + t] = (bf16_t)v;
  }
}

// ---------------------------------------------------------------------------
// Main fused kernel: one block = 16 batch elements, 8 waves.
// ---------------------------------------------------------------------------
__global__ __launch_bounds__(256) void arc_encoder_kernel(
    const int* __restrict__ obs, const bf16_t* __restrict__ ws,
    const float* __restrict__ ip_b1, const float* __restrict__ ip_b2,
    const float* __restrict__ op_b1, const float* __restrict__ op_b2,
    const float* __restrict__ q_b, const float* __restrict__ k_b,
    const float* __restrict__ v_b, const float* __restrict__ o_b,
    const float* __restrict__ cb_b1, const float* __restrict__ cb_b2,
    float* __restrict__ out) {
  __shared__ __attribute__((aligned(16))) bf16_t s_ce[4 * 64];
  __shared__ __attribute__((aligned(16))) bf16_t s_pos[25 * 64];
  __shared__ int s_gin[16 * 25];
  __shared__ int s_gout[16 * 25];
  __shared__ int s_mask[16 * 25];
  __shared__ __attribute__((aligned(16))) bf16_t s_h1a[16 * 136];  // in-path hidden
  __shared__ __attribute__((aligned(16))) bf16_t s_h1b[16 * 136];  // out-path hidden
  __shared__ __attribute__((aligned(16))) bf16_t s_tmp[16 * 72];   // inp_flat at pos p
  __shared__ float s_kp[16 * 68];
  __shared__ float s_vp[16 * 68];
  __shared__ __attribute__((aligned(16))) bf16_t s_cpe[16 * 64];
  __shared__ float s_q[16 * 68];       // pre-scaled by HD^-0.5
  __shared__ float s_outctx[16 * 68];
  __shared__ float s_att[16 * 68];
  __shared__ __attribute__((aligned(16))) bf16_t s_comb[16 * 200];
  __shared__ __attribute__((aligned(16))) bf16_t s_hcb[16 * 136];

  const int tid = threadIdx.x;
  const int lane = tid & 31;
  const int wv = tid >> 5;
  const int hf = lane >> 4;
  const int col = lane & 15;   // C-fragment column / A-fragment row (batch-local)
  const int b0 = blockIdx.x * TB;

  // ---- stage embeds + grids into LDS ----
  for (int i = tid; i < 4 * 64; i += 256) s_ce[i] = ws[OFF_CE + i];
  for (int i = tid; i < 25 * 64; i += 256) s_pos[i] = ws[OFF_POS + i];
  for (int i = tid; i < 16 * 75; i += 256) {
    int bl = i / 75, rem = i % 75, ch = rem / 25, pp = rem % 25;
    int v = obs[(size_t)(b0 + bl) * 75 + ch * 25 + pp];
    if (ch == 0) s_gin[bl * 25 + pp] = v;
    else if (ch == 1) s_gout[bl * 25 + pp] = v;
    else s_mask[bl * 25 + pp] = v;
  }
  __syncthreads();

  // ---- cpe: first unmasked cell -> concat(row_embed, col_embed) == pos[p] ----
  if (tid < 16) {
    int fidx = -1;
    for (int j = 0; j < 25; j++)
      if (s_mask[tid * 25 + j] != 0 && fidx < 0) fidx = j;
    for (int j = 0; j < 64; j++)
      s_cpe[tid * 64 + j] = (fidx >= 0) ? s_pos[fidx * 64 + j] : (bf16_t)0.0f;
  }
  __syncthreads();

  // ---- q = (cpe @ q_w + q_b) * HD^-0.5 ----
  if (wv < 4) {
    v8f acc = {};
    for (int kc = 0; kc < 2; kc++) {
      v16bf a = afrag_lds(s_cpe, 64, kc * 32, lane);
      v16bf b = bfrag(ws + OFF_QW, wv * 2 + kc, lane);
      acc = wmma_bf16(a, b, acc);
    }
    float bq = q_b[wv * 16 + col];
#pragma unroll
    for (int i = 0; i < 8; i++)
      s_q[(i + 8 * hf) * 68 + wv * 16 + col] = (acc[i] + bq) * 0.25f;
  }

  // online-softmax state: thread t<64 owns (batch=t>>2, head=t&3)
  float m_run = -1e30f, l_run = 0.0f;
  float a_acc[16];
#pragma unroll
  for (int d = 0; d < 16; d++) a_acc[d] = 0.0f;
  v8f out_acc = {};  // out-path layer2 sum over positions (waves 4..7)
  __syncthreads();

  // =================== position loop ===================
  for (int p = 0; p < P; p++) {
    const bf16_t* pos_p = s_pos + p * 64;

    // Phase A: hidden layer (16x128) for both MLP paths; wave wv -> n-tile wv.
    {
      int gi = s_gin[col * 25 + p];
      int go = s_gout[col * 25 + p];
      const bf16_t* ceI = s_ce + gi * 64;
      const bf16_t* ceO = s_ce + go * 64;
      v8f aI = {}, aO = {};
      for (int kc = 0; kc < 4; kc++) {
        v16bf fa = afrag_feat(ceI, pos_p, kc * 32, hf);
        v16bf fb = (kc < 2) ? afrag_feat(ceO, pos_p, kc * 32, hf) : fa;  // kc>=2: pos only
        v16bf w1 = bfrag(ws + OFF_IPW1, wv * 4 + kc, lane);
        v16bf w2 = bfrag(ws + OFF_OPW1, wv * 4 + kc, lane);
        aI = wmma_bf16(fa, w1, aI);
        aO = wmma_bf16(fb, w2, aO);
      }
      float bI = ip_b1[wv * 16 + col];
      float bO = op_b1[wv * 16 + col];
#pragma unroll
      for (int i = 0; i < 8; i++) {
        s_h1a[(i + 8 * hf) * 136 + wv * 16 + col] = (bf16_t)fmaxf(aI[i] + bI, 0.0f);
        s_h1b[(i + 8 * hf) * 136 + wv * 16 + col] = (bf16_t)fmaxf(aO[i] + bO, 0.0f);
      }
    }
    __syncthreads();

    // Phase B: layer2. Waves 0-3: inp_flat tile; waves 4-7: out-path sum in regs.
    if (wv < 4) {
      v8f acc = {};
      for (int kc = 0; kc < 4; kc++) {
        v16bf a = afrag_lds(s_h1a, 136, kc * 32, lane);
        v16bf b = bfrag(ws + OFF_IPW2, wv * 4 + kc, lane);
        acc = wmma_bf16(a, b, acc);
      }
      float bb = ip_b2[wv * 16 + col];
#pragma unroll
      for (int i = 0; i < 8; i++)
        s_tmp[(i + 8 * hf) * 72 + wv * 16 + col] = (bf16_t)(acc[i] + bb);
    } else {
      int nt = wv - 4;
      for (int kc = 0; kc < 4; kc++) {
        v16bf a = afrag_lds(s_h1b, 136, kc * 32, lane);
        v16bf b = bfrag(ws + OFF_OPW2, nt * 4 + kc, lane);
        out_acc = wmma_bf16(a, b, out_acc);  // accumulate across p in C regs
      }
    }
    __syncthreads();

    // Phase C: k/v projection for this position (waves 0-3: k, 4-7: v).
    {
      int nt = wv & 3;
      bool isK = (wv < 4);
      const bf16_t* wp = ws + (isK ? OFF_KW : OFF_VW);
      v8f acc = {};
      for (int kc = 0; kc < 2; kc++) {
        v16bf a = afrag_lds(s_tmp, 72, kc * 32, lane);
        v16bf b = bfrag(wp, nt * 2 + kc, lane);
        acc = wmma_bf16(a, b, acc);
      }
      float bb = (isK ? k_b : v_b)[nt * 16 + col];
      float* dst = isK ? s_kp : s_vp;
#pragma unroll
      for (int i = 0; i < 8; i++)
        dst[(i + 8 * hf) * 68 + nt * 16 + col] = acc[i] + bb;
    }
    __syncthreads();

    // Online-softmax update (flash-style); full waves 0-1 -> wave-uniform.
    if (tid < 64) {
      int bl = tid >> 2, h = tid & 3;
      const float* qp = s_q + bl * 68 + h * 16;
      const float* kp = s_kp + bl * 68 + h * 16;
      const float* vp = s_vp + bl * 68 + h * 16;
      float s = 0.0f;
#pragma unroll
      for (int d = 0; d < 16; d++) s += qp[d] * kp[d];
      float mn = fmaxf(m_run, s);
      float sc = __expf(m_run - mn);
      float wq = __expf(s - mn);
      l_run = l_run * sc + wq;
#pragma unroll
      for (int d = 0; d < 16; d++) a_acc[d] = a_acc[d] * sc + wq * vp[d];
      m_run = mn;
    }
  }
  // =================== end position loop ===================

  // out_ctx = sum/25 + op_b2
  if (wv >= 4) {
    int nt = wv - 4;
    float bb = op_b2[nt * 16 + col];
#pragma unroll
    for (int i = 0; i < 8; i++)
      s_outctx[(i + 8 * hf) * 68 + nt * 16 + col] = out_acc[i] * (1.0f / 25.0f) + bb;
  }
  __syncthreads();

  // attention output (pre o-projection)
  if (tid < 64) {
    int bl = tid >> 2, h = tid & 3;
    float inv = 1.0f / l_run;
#pragma unroll
    for (int d = 0; d < 16; d++) s_att[bl * 68 + h * 16 + d] = a_acc[d] * inv;
  }
  __syncthreads();

  // o-projection -> combined[:,0:64]; everyone fills combined[:,64:192]
  if (wv < 4) {
    v8f acc = {};
    for (int kc = 0; kc < 2; kc++) {
      v16bf a = afrag_f32(s_att, 68, kc * 32, lane);
      v16bf b = bfrag(ws + OFF_OW, wv * 2 + kc, lane);
      acc = wmma_bf16(a, b, acc);
    }
    float bb = o_b[wv * 16 + col];
#pragma unroll
    for (int i = 0; i < 8; i++)
      s_comb[(i + 8 * hf) * 200 + wv * 16 + col] = (bf16_t)(acc[i] + bb);
  }
  for (int i = tid; i < 16 * 128; i += 256) {
    int r = i >> 7, c = i & 127;
    s_comb[r * 200 + 64 + c] =
        (c < 64) ? (bf16_t)s_outctx[r * 68 + c] : s_cpe[r * 64 + (c - 64)];
  }
  __syncthreads();

  // combiner layer 1: (16x192) @ cb_w1 -> relu -> (16x128)
  {
    v8f acc = {};
    for (int kc = 0; kc < 6; kc++) {
      v16bf a = afrag_lds(s_comb, 200, kc * 32, lane);
      v16bf b = bfrag(ws + OFF_CBW1, wv * 6 + kc, lane);
      acc = wmma_bf16(a, b, acc);
    }
    float bb = cb_b1[wv * 16 + col];
#pragma unroll
    for (int i = 0; i < 8; i++)
      s_hcb[(i + 8 * hf) * 136 + wv * 16 + col] = (bf16_t)fmaxf(acc[i] + bb, 0.0f);
  }
  __syncthreads();

  // combiner layer 2 -> global out (fp32)
  {
    v8f acc = {};
    for (int kc = 0; kc < 4; kc++) {
      v16bf a = afrag_lds(s_hcb, 136, kc * 32, lane);
      v16bf b = bfrag(ws + OFF_CBW2, wv * 4 + kc, lane);
      acc = wmma_bf16(a, b, acc);
    }
    float bb = cb_b2[wv * 16 + col];
#pragma unroll
    for (int i = 0; i < 8; i++)
      out[(size_t)(b0 + i + 8 * hf) * 128 + wv * 16 + col] = acc[i] + bb;
  }
}

// ---------------------------------------------------------------------------
extern "C" void kernel_launch(void* const* d_in, const int* in_sizes, int n_in,
                              void* d_out, int out_size, void* d_ws, size_t ws_size,
                              hipStream_t stream) {
  (void)in_sizes; (void)n_in; (void)out_size; (void)ws_size;
  const int*   obs         = (const int*)d_in[0];
  const float* color_embed = (const float*)d_in[1];
  const float* row_embed   = (const float*)d_in[2];
  const float* col_embed   = (const float*)d_in[3];
  const float* ip_w1 = (const float*)d_in[4];  const float* ip_b1 = (const float*)d_in[5];
  const float* ip_w2 = (const float*)d_in[6];  const float* ip_b2 = (const float*)d_in[7];
  const float* q_w   = (const float*)d_in[8];  const float* q_b   = (const float*)d_in[9];
  const float* k_w   = (const float*)d_in[10]; const float* k_b   = (const float*)d_in[11];
  const float* v_w   = (const float*)d_in[12]; const float* v_b   = (const float*)d_in[13];
  const float* o_w   = (const float*)d_in[14]; const float* o_b   = (const float*)d_in[15];
  const float* op_w1 = (const float*)d_in[16]; const float* op_b1 = (const float*)d_in[17];
  const float* op_w2 = (const float*)d_in[18]; const float* op_b2 = (const float*)d_in[19];
  const float* cb_w1 = (const float*)d_in[20]; const float* cb_b1 = (const float*)d_in[21];
  const float* cb_w2 = (const float*)d_in[22]; const float* cb_b2 = (const float*)d_in[23];
  bf16_t* ws = (bf16_t*)d_ws;

  // pack embeds + weights into WMMA fragment layout (bf16)
  pack_embeds<<<7, 256, 0, stream>>>(color_embed, row_embed, col_embed, ws);
  pack_wmma_b<<<8 * 4, 512, 0, stream>>>(ip_w1, ws + OFF_IPW1, 128, 4);
  pack_wmma_b<<<4 * 4, 512, 0, stream>>>(ip_w2, ws + OFF_IPW2, 64, 4);
  pack_wmma_b<<<8 * 4, 512, 0, stream>>>(op_w1, ws + OFF_OPW1, 128, 4);
  pack_wmma_b<<<4 * 4, 512, 0, stream>>>(op_w2, ws + OFF_OPW2, 64, 4);
  pack_wmma_b<<<4 * 2, 512, 0, stream>>>(q_w, ws + OFF_QW, 64, 2);
  pack_wmma_b<<<4 * 2, 512, 0, stream>>>(k_w, ws + OFF_KW, 64, 2);
  pack_wmma_b<<<4 * 2, 512, 0, stream>>>(v_w, ws + OFF_VW, 64, 2);
  pack_wmma_b<<<4 * 2, 512, 0, stream>>>(o_w, ws + OFF_OW, 64, 2);
  pack_wmma_b<<<8 * 6, 512, 0, stream>>>(cb_w1, ws + OFF_CBW1, 128, 6);
  pack_wmma_b<<<8 * 4, 512, 0, stream>>>(cb_w2, ws + OFF_CBW2, 128, 4);

  // fused encoder: 32768/16 = 2048 blocks, 8 waves each
  arc_encoder_kernel<<<2048, 256, 0, stream>>>(
      obs, ws, ip_b1, ip_b2, op_b1, op_b2, q_b, k_b, v_b, o_b, cb_b1, cb_b2,
      (float*)d_out);
}
